// Phi3SmallSelfAttention_18365280158360
// MI455X (gfx1250) — matrix operand order repaired and checked by reading, not compiled, not run on hardware
//
#include <hip/hip_runtime.h>
#include <math.h>

// ---------------------------------------------------------------------------
// Phi3-Small block-sparse self-attention for gfx1250 (MI455X), wave32 + WMMA.
// Matmuls via v_wmma_f32_16x16x32_f16; attention K tiles staged by the
// Tensor Data Mover (tensor_load_to_lds + s_wait_tensorcnt).
// ---------------------------------------------------------------------------

typedef _Float16 v4h  __attribute__((ext_vector_type(4)));
typedef _Float16 v8h  __attribute__((ext_vector_type(8)));
typedef _Float16 v16h __attribute__((ext_vector_type(16)));
typedef float    v8f  __attribute__((ext_vector_type(8)));
typedef unsigned int v4u __attribute__((ext_vector_type(4)));
typedef int      v4i  __attribute__((ext_vector_type(4)));
typedef int      v8i  __attribute__((ext_vector_type(8)));

#define T_TOK 2048
#define NHEAD 32
#define NKV   8
#define HD    128
#define DMODEL 4096
#define BS    64
#define LOCAL 16
#define VERT  8
#define ATT_SCALE 0.08838834764831845f   // 1/sqrt(128)

// Assemble a 16-wide f16 fragment from two aligned 8-wide LDS/global chunks.
__device__ __forceinline__ v16h load_frag16(const _Float16* p0, const _Float16* p1) {
    v8h lo = *(const v8h*)p0;
    v8h hi = *(const v8h*)p1;
    return __builtin_shufflevector(lo, hi, 0,1,2,3,4,5,6,7,8,9,10,11,12,13,14,15);
}

// ---------------------------------------------------------------------------
// Tiled GEMM: C[M,N] = A[M,K] (fp32) @ B[K,N] (fp32) + bias, fp32 out.
// Block tile 128x128x32; 256 threads = 8 waves; each wave: 2x4 = 8 WMMA tiles.
// fp32 -> f16 conversion happens while staging into LDS; register
// double-buffering overlaps global loads with the WMMA stream.
// ---------------------------------------------------------------------------
#define LDSA_STRIDE 40   // 32 K + 8 pad (f16), 80B rows -> 16B aligned
__global__ __launch_bounds__(256)
void gemm_f16wmma_kernel(const float* __restrict__ A, const float* __restrict__ B,
                         const float* __restrict__ bias, float* __restrict__ C,
                         int M, int N, int K) {
    __shared__ __align__(16) _Float16 Alds[128 * LDSA_STRIDE];
    __shared__ __align__(16) _Float16 Blds[128 * LDSA_STRIDE];

    const int tid  = threadIdx.x;
    const int lane = tid & 31;
    const int wave = tid >> 5;      // 0..7
    const int wm   = wave >> 1;     // 0..3 -> 32-row group
    const int wn   = wave & 1;      // 0..1 -> 64-col group
    const int half = lane >> 4;
    const int l15  = lane & 15;
    const int row0 = blockIdx.y * 128;
    const int col0 = blockIdx.x * 128;

    v8f acc[2][4];
    const v8f zero8 = {0.f,0.f,0.f,0.f,0.f,0.f,0.f,0.f};
    for (int i = 0; i < 2; ++i)
        for (int j = 0; j < 4; ++j)
            acc[i][j] = zero8;

    float4 ra[4], rb[4];

    // Per-thread staging coordinates: i = tid*4 + u*1024 covers 0..4095.
    #define GEMM_LOAD(K0)                                                      \
        _Pragma("unroll")                                                      \
        for (int u = 0; u < 4; ++u) {                                          \
            int i = tid * 4 + u * 1024;                                        \
            int r = i >> 5, c = i & 31;                                        \
            ra[u] = *(const float4*)&A[(size_t)(row0 + r) * K + (K0) + c];     \
            int kk = i >> 7, cc = i & 127;                                     \
            rb[u] = *(const float4*)&B[(size_t)((K0) + kk) * N + col0 + cc];   \
        }

    #define GEMM_STORE()                                                       \
        _Pragma("unroll")                                                      \
        for (int u = 0; u < 4; ++u) {                                          \
            int i = tid * 4 + u * 1024;                                        \
            int r = i >> 5, c = i & 31;                                        \
            v4h ah = {(_Float16)ra[u].x, (_Float16)ra[u].y,                    \
                      (_Float16)ra[u].z, (_Float16)ra[u].w};                   \
            *(v4h*)&Alds[r * LDSA_STRIDE + c] = ah;                            \
            int kk = i >> 7, cc = i & 127;                                     \
            Blds[(cc + 0) * LDSA_STRIDE + kk] = (_Float16)rb[u].x;             \
            Blds[(cc + 1) * LDSA_STRIDE + kk] = (_Float16)rb[u].y;             \
            Blds[(cc + 2) * LDSA_STRIDE + kk] = (_Float16)rb[u].z;             \
            Blds[(cc + 3) * LDSA_STRIDE + kk] = (_Float16)rb[u].w;             \
        }

    GEMM_LOAD(0);
    GEMM_STORE();

    for (int k0 = 0; k0 < K; k0 += 32) {
        __syncthreads();                       // LDS tile k0 visible
        const bool more = (k0 + 32 < K);
        if (more) { GEMM_LOAD(k0 + 32); }      // overlap with WMMAs below
        if (k0 + 64 < K) {                     // global_prefetch_b8 for k0+64
            __builtin_prefetch(&A[(size_t)(row0 + (tid >> 1)) * K + (k0 + 64) + (tid & 1) * 16], 0, 1);
            __builtin_prefetch(&B[(size_t)(k0 + 64 + (tid >> 3)) * N + col0 + (tid & 7) * 16], 0, 1);
        }

        v16h afrag[2];
        #pragma unroll
        for (int i = 0; i < 2; ++i) {
            const _Float16* p = &Alds[(wm * 32 + i * 16 + l15) * LDSA_STRIDE];
            afrag[i] = load_frag16(p + 8 * half, p + 16 + 8 * half);
        }
        #pragma unroll
        for (int j = 0; j < 4; ++j) {
            const _Float16* p =
                &Blds[(wn * 64 + j * 16 + l15) * LDSA_STRIDE + 8 * half];
            v16h bfrag = load_frag16(p, p + 16);
            #pragma unroll
            for (int i = 0; i < 2; ++i) {
                acc[i][j] = __builtin_amdgcn_wmma_f32_16x16x32_f16(
                    false, afrag[i], false, bfrag, (short)0, acc[i][j],
                    false, false);
            }
        }
        __syncthreads();                       // done reading tile k0
        if (more) { GEMM_STORE(); }            // publish tile k0+32
    }

    // Epilogue: C layout -> VGPR r holds row (8*half + r), col = l15.
    #pragma unroll
    for (int i = 0; i < 2; ++i) {
        #pragma unroll
        for (int j = 0; j < 4; ++j) {
            int col = col0 + wn * 64 + j * 16 + l15;
            float bv = bias ? bias[col] : 0.f;
            #pragma unroll
            for (int r = 0; r < 8; ++r) {
                int row = row0 + wm * 32 + i * 16 + half * 8 + r;
                C[(size_t)row * N + col] = acc[i][j][r] + bv;
            }
        }
    }
    #undef GEMM_LOAD
    #undef GEMM_STORE
}

// ---------------------------------------------------------------------------
// RoPE + fp32->f16 repack. Q: [T, H, HD] fp32 -> [H, T, HD] f16 (head-major).
// ---------------------------------------------------------------------------
__global__ __launch_bounds__(64)
void rope_q_kernel(const float* __restrict__ qf, const int* __restrict__ pos,
                   _Float16* __restrict__ qh) {
    const int t = blockIdx.x, h = blockIdx.y, d = threadIdx.x; // d in [0,64)
    float e  = -((float)(2 * d) / (float)HD) * 19.93156857f;   // log2(1e6)
    float fr = (float)pos[t] * exp2f(e);
    float c = __cosf(fr), s = __sinf(fr);
    const float* q = qf + (size_t)t * DMODEL + h * HD;
    float x1 = q[d], x2 = q[d + 64];
    _Float16* o = qh + ((size_t)h * T_TOK + t) * HD;
    o[d]      = (_Float16)(x1 * c - x2 * s);
    o[d + 64] = (_Float16)(x2 * c + x1 * s);
}

__global__ __launch_bounds__(64)
void rope_kv_kernel(const float* __restrict__ kf, const float* __restrict__ vf,
                    const int* __restrict__ pos,
                    _Float16* __restrict__ kh, _Float16* __restrict__ vh) {
    const int t = blockIdx.x, h = blockIdx.y, d = threadIdx.x;
    float e  = -((float)(2 * d) / (float)HD) * 19.93156857f;
    float fr = (float)pos[t] * exp2f(e);
    float c = __cosf(fr), s = __sinf(fr);
    const float* k = kf + (size_t)t * (NKV * HD) + h * HD;
    const float* v = vf + (size_t)t * (NKV * HD) + h * HD;
    _Float16* ok = kh + ((size_t)h * T_TOK + t) * HD;
    _Float16* ov = vh + ((size_t)h * T_TOK + t) * HD;
    float x1 = k[d], x2 = k[d + 64];
    ok[d]      = (_Float16)(x1 * c - x2 * s);
    ok[d + 64] = (_Float16)(x2 * c + x1 * s);
    ov[d]      = (_Float16)v[d];
    ov[d + 64] = (_Float16)v[d + 64];
}

// ---------------------------------------------------------------------------
// Block-sparse flash attention. Grid: (T/64 q-blocks, 32 heads), 128 threads.
// K tile staged by the Tensor Data Mover (64x128 f16, LDS pad 4 DWORDs per
// 64-DWORD row -> 136-half padded rows); V staged manually transposed.
// ---------------------------------------------------------------------------
#define KSTRIDE 136   // 128 + 8 pad (f16) -> 272B rows, 16B aligned
#define PSTRIDE 72    // 64 + 8 pad
__global__ __launch_bounds__(128)
void attn_sparse_kernel(const _Float16* __restrict__ qh,
                        const _Float16* __restrict__ kh,
                        const _Float16* __restrict__ vh,
                        float* __restrict__ ctx) {
    __shared__ __align__(16) _Float16 Klds[64 * KSTRIDE];   // [key][hd]
    __shared__ __align__(16) _Float16 Vtlds[128 * PSTRIDE]; // [hd][key]
    __shared__ __align__(16) _Float16 Plds[64 * PSTRIDE];   // [qrow][key]

    const int qb   = blockIdx.x;
    const int h    = blockIdx.y;
    const int hkv  = h >> 2;               // H/HKV = 4
    const int tid  = threadIdx.x;
    const int lane = tid & 31;
    const int wave = tid >> 5;             // 0..3
    const int half = lane >> 4;
    const int l15  = lane & 15;

    // Q fragments: 16 rows x 128 (4 k-steps of 32), loaded once from global.
    v16h qf[4];
    {
        int row = qb * BS + wave * 16 + l15;
        const _Float16* qp = qh + ((size_t)h * T_TOK + row) * HD;
        #pragma unroll
        for (int s = 0; s < 4; ++s)
            qf[s] = load_frag16(qp + 32 * s + 8 * half,
                                qp + 32 * s + 16 + 8 * half);
    }

    const v8f zero8 = {0.f,0.f,0.f,0.f,0.f,0.f,0.f,0.f};
    v8f o[8];
    #pragma unroll
    for (int j = 0; j < 8; ++j) o[j] = zero8;
    float mrow[8], lrow[8];
    #pragma unroll
    for (int r = 0; r < 8; ++r) { mrow[r] = -3.0e38f; lrow[r] = 0.f; }

    const unsigned klds_base = (unsigned)(size_t)(&Klds[0]);  // LDS byte offset

    for (int kb = 0; kb <= qb; ++kb) {
        // Uniform per-block sparsity decision (no EXEC divergence).
        bool allowed = ((qb - kb) < LOCAL) || (((kb + h + 1) & (VERT - 1)) == 0);
        if (!allowed) continue;

        __syncthreads();   // previous iteration done reading K/V LDS

        const _Float16* kbase = kh + ((size_t)hkv * T_TOK + kb * BS) * HD;
        const _Float16* vbase = vh + ((size_t)hkv * T_TOK + kb * BS) * HD;

        // ---- TDM: DMA the 64x128 f16 K tile into LDS with row padding -----
        // D# group0: count=1 | lds_addr | global_addr | type=2
        // D# group1: data_size=2B, pad_enable, pad_interval=64 DW (code 5),
        //            pad_amount=4 DW (code 3) -> 136-half LDS rows;
        //            tensor_dim0=128, tensor_dim1=64, tile 128x64, stride 128.
        if (wave == 0) {
            unsigned long long ga = (unsigned long long)(size_t)kbase;
            v4u g0 = { 1u, klds_base, (unsigned)ga,
                       ((unsigned)(ga >> 32) & 0x01FFFFFFu) | 0x80000000u };
            v8i g1 = { (int)((1u << 16) | (1u << 20) | (5u << 22) | (3u << 25)),
                       (int)(128u << 16),   // tensor_dim0 = 128
                       (int)(64u  << 16),   // tensor_dim1 = 64
                       (int)(128u << 16),   // tile_dim0 = 128
                       64,                  // tile_dim1 = 64
                       128,                 // tensor_dim0_stride = 128
                       0, 0 };
            v4i gz = { 0, 0, 0, 0 };
#if __clang_major__ >= 23
            v8i gz8 = { 0, 0, 0, 0, 0, 0, 0, 0 };
            __builtin_amdgcn_tensor_load_to_lds(g0, g1, gz, gz, gz8, 0);
#else
            __builtin_amdgcn_tensor_load_to_lds(g0, g1, gz, gz, 0);
#endif
        }

        // ---- stage V transposed (vectorized v8h loads, b16 scatter) -------
        #pragma unroll
        for (int it = 0; it < 8; ++it) {
            int i = tid * 8 + it * 1024;    // covers 0..8191
            int r = i >> 7, c = i & 127;
            v8h x = *(const v8h*)&vbase[(size_t)r * HD + c];
            #pragma unroll
            for (int j = 0; j < 8; ++j)
                Vtlds[(c + j) * PSTRIDE + r] = x[j];
        }

        if (wave == 0) __builtin_amdgcn_s_wait_tensorcnt(0);
        __syncthreads();   // K (TDM) + Vt visible to all waves

        // ---- S = Q K^T : 4 key tiles x 4 k-steps --------------------------
        v8f s[4];
        #pragma unroll
        for (int t = 0; t < 4; ++t) s[t] = zero8;
        #pragma unroll
        for (int t = 0; t < 4; ++t) {
            #pragma unroll
            for (int ks = 0; ks < 4; ++ks) {
                const _Float16* p =
                    &Klds[(t * 16 + l15) * KSTRIDE + 32 * ks + 8 * half];
                v16h bf = load_frag16(p, p + 16);
                s[t] = __builtin_amdgcn_wmma_f32_16x16x32_f16(
                    false, qf[ks], false, bf, (short)0, s[t], false, false);
            }
        }

        // ---- scale + causal mask (diagonal block only) --------------------
        const bool diag = (kb == qb);
        #pragma unroll
        for (int t = 0; t < 4; ++t) {
            #pragma unroll
            for (int r = 0; r < 8; ++r) {
                float v = s[t][r] * ATT_SCALE;
                int qrow = wave * 16 + half * 8 + r;
                int kcol = t * 16 + l15;
                if (diag && (kcol > qrow)) v = -3.0e38f;
                s[t][r] = v;
            }
        }

        // ---- online softmax (16-lane reductions within each half) ---------
        #pragma unroll
        for (int r = 0; r < 8; ++r) {
            float mx = fmaxf(fmaxf(s[0][r], s[1][r]), fmaxf(s[2][r], s[3][r]));
            #pragma unroll
            for (int off = 1; off < 16; off <<= 1)
                mx = fmaxf(mx, __shfl_xor(mx, off, 32));
            float mnew  = fmaxf(mrow[r], mx);
            float alpha = __expf(mrow[r] - mnew);
            mrow[r] = mnew;
            float rs = 0.f;
            #pragma unroll
            for (int t = 0; t < 4; ++t) {
                float p = __expf(s[t][r] - mnew);
                s[t][r] = p;
                rs += p;
            }
            #pragma unroll
            for (int off = 1; off < 16; off <<= 1)
                rs += __shfl_xor(rs, off, 32);
            lrow[r] = lrow[r] * alpha + rs;
            #pragma unroll
            for (int j = 0; j < 8; ++j) o[j][r] *= alpha;
        }

        // ---- P (C-layout) -> LDS -> A-layout fragments --------------------
        #pragma unroll
        for (int t = 0; t < 4; ++t)
            #pragma unroll
            for (int r = 0; r < 8; ++r)
                Plds[(wave * 16 + half * 8 + r) * PSTRIDE + t * 16 + l15] =
                    (_Float16)s[t][r];
        __syncthreads();

        // ---- O += P V : 8 hd tiles x 2 k-steps of 32 ----------------------
        #pragma unroll
        for (int ks = 0; ks < 2; ++ks) {
            const _Float16* pa = &Plds[(wave * 16 + l15) * PSTRIDE + 32 * ks];
            v16h af = load_frag16(pa + 8 * half, pa + 16 + 8 * half);
            #pragma unroll
            for (int j = 0; j < 8; ++j) {
                const _Float16* pb =
                    &Vtlds[(j * 16 + l15) * PSTRIDE + 32 * ks + 8 * half];
                v16h bf = load_frag16(pb, pb + 16);
                o[j] = __builtin_amdgcn_wmma_f32_16x16x32_f16(
                    false, af, false, bf, (short)0, o[j], false, false);
            }
        }
    }

    // ---- normalize + write ctx[t][h*HD + d] fp32 --------------------------
    #pragma unroll
    for (int r = 0; r < 8; ++r) {
        float inv = 1.0f / lrow[r];
        int row = qb * BS + wave * 16 + half * 8 + r;
        float* op = ctx + (size_t)row * DMODEL + h * HD;
        #pragma unroll
        for (int j = 0; j < 8; ++j)
            op[j * 16 + l15] = o[j][r] * inv;
    }
}

// ---------------------------------------------------------------------------
// Host-side orchestration.
// ---------------------------------------------------------------------------
extern "C" void kernel_launch(void* const* d_in, const int* in_sizes, int n_in,
                              void* d_out, int out_size, void* d_ws, size_t ws_size,
                              hipStream_t stream) {
    const float* hidden = (const float*)d_in[0];
    const int*   pos    = (const int*)  d_in[1];
    const float* Wq     = (const float*)d_in[2];
    const float* bq     = (const float*)d_in[3];
    const float* Wk     = (const float*)d_in[4];
    const float* bk     = (const float*)d_in[5];
    const float* Wv     = (const float*)d_in[6];
    const float* bv     = (const float*)d_in[7];
    const float* Wo     = (const float*)d_in[8];
    const float* bo     = (const float*)d_in[9];
    float*       out    = (float*)d_out;

    char* ws = (char*)d_ws;
    size_t off = 0;
    float* qf = (float*)(ws + off); off += (size_t)T_TOK * DMODEL * 4;
    float* kf = (float*)(ws + off); off += (size_t)T_TOK * NKV * HD * 4;
    float* vf = (float*)(ws + off); off += (size_t)T_TOK * NKV * HD * 4;
    _Float16* qh = (_Float16*)(ws + off); off += (size_t)NHEAD * T_TOK * HD * 2;
    _Float16* kh = (_Float16*)(ws + off); off += (size_t)NKV  * T_TOK * HD * 2;
    _Float16* vh = (_Float16*)(ws + off); off += (size_t)NKV  * T_TOK * HD * 2;
    float* ctx = (float*)(ws + off); off += (size_t)T_TOK * DMODEL * 4;

    // 1) QKV projections (WMMA GEMMs)
    gemm_f16wmma_kernel<<<dim3(DMODEL / 128, T_TOK / 128), 256, 0, stream>>>(
        hidden, Wq, bq, qf, T_TOK, DMODEL, DMODEL);
    gemm_f16wmma_kernel<<<dim3((NKV * HD) / 128, T_TOK / 128), 256, 0, stream>>>(
        hidden, Wk, bk, kf, T_TOK, NKV * HD, DMODEL);
    gemm_f16wmma_kernel<<<dim3((NKV * HD) / 128, T_TOK / 128), 256, 0, stream>>>(
        hidden, Wv, bv, vf, T_TOK, NKV * HD, DMODEL);

    // 2) RoPE + f16 repack
    rope_q_kernel<<<dim3(T_TOK, NHEAD), 64, 0, stream>>>(qf, pos, qh);
    rope_kv_kernel<<<dim3(T_TOK, NKV), 64, 0, stream>>>(kf, vf, pos, kh, vh);

    // 3) Block-sparse flash attention (TDM-staged K tiles)
    attn_sparse_kernel<<<dim3(T_TOK / BS, NHEAD), 128, 0, stream>>>(qh, kh, vh, ctx);

    // 4) Output projection
    gemm_f16wmma_kernel<<<dim3(DMODEL / 128, T_TOK / 128), 256, 0, stream>>>(
        ctx, Wo, bo, out, T_TOK, DMODEL, DMODEL);
}
